// RNN_83073257439559
// MI455X (gfx1250) — matrix operand order repaired and checked by reading, not compiled, not run on hardware
//
#include <hip/hip_runtime.h>
#include <hip/hip_fp16.h>

typedef __attribute__((ext_vector_type(16))) _Float16 v16h;
typedef __attribute__((ext_vector_type(8)))  _Float16 v8h;
typedef __attribute__((ext_vector_type(8)))  float    v8f;

#define WMMA_F16(a, b, c) \
  __builtin_amdgcn_wmma_f32_16x16x32_f16(false, (a), false, (b), (short)0, (c), false, false)

static constexpr int Bsz = 64;     // batch
static constexpr int T   = 2048;   // timesteps
static constexpr int F   = 64;     // input features
static constexpr int H   = 128;    // rnn size
static constexpr int G   = 4 * H;  // 512 gate columns (i,f,g,o)
static constexpr int TILE = 512;   // f16 elements per 16x32 fragment tile (1 KB)

__device__ __forceinline__ float fast_sigmoid(float x) {
  return __builtin_amdgcn_rcpf(1.0f + __expf(-x));   // v_exp_f32 + v_rcp_f32
}
__device__ __forceinline__ float fast_tanh(float x) {
  return 1.0f - 2.0f * __builtin_amdgcn_rcpf(__expf(2.0f * x) + 1.0f);
}

// Swizzled position (f16 units) of B-matrix element (k,n) inside a 32x16 tile.
// Lane = n + 16*((k>>3)&1); element = (k&7) + 8*(k>=16)  (same k order as a_swz).
__device__ __forceinline__ int b_swz(int k, int n) {
  return (n + 16 * ((k >> 3) & 1)) * 16 + (k & 7) + ((k & 16) ? 8 : 0);
}

// Swizzled position (f16 units) of A-matrix element (m,k) inside a 16x32 tile.
// Lanes 0-15 hold K 0-7 & 16-23; lanes 16-31 hold K 8-15 & 24-31.
__device__ __forceinline__ int a_swz(int m, int k) {
  const int half = (k >> 3) & 1;
  const int e    = (k & 7) + ((k & 16) ? 8 : 0);
  return (m + 16 * half) * 16 + e;
}

// Fragment load: lane's 32 contiguous bytes -> 2x ds_load_b128 / global_load_b128.
__device__ __forceinline__ v16h load_frag(const _Float16* tile, int lane) {
  return *(const v16h*)(tile + lane * 16);
}

// ---------------------------------------------------------------------------
// Kernel 1: fused input-projection + LSTM recurrence.
// grid = 4 (one WG per 16-batch group), block = 256 (8 waves).
// Wave w owns hidden columns [16w,16w+16): 24-WMMA chain per step (4 gate
// accumulators interleaved), B-fragment loads software-pipelined 2 deep so no
// WMMA waits on a full LDS drain. One barrier per step (double-buffered
// h / x tiles). First WMMA of each accumulator uses inline C = 0.
// hs (global) layout: [group][t][kt0..3][512 f16]  (A-fragment tile blocks).
// ---------------------------------------------------------------------------
__global__ __launch_bounds__(256) void lstm_kernel(
    const float* __restrict__ x, const float* __restrict__ W,
    const float* __restrict__ U, const float* __restrict__ b,
    _Float16* __restrict__ hs) {
  extern __shared__ char smem[];
  _Float16* Wsw = (_Float16*)smem;            // 2 kt x 32 nt tiles  (64 KB)
  _Float16* Usw = Wsw + 2 * 32 * TILE;        // 4 kt x 32 nt tiles  (128 KB)
  _Float16* hb0 = Usw + 4 * 32 * TILE;        // h tile buffer 0      (4 KB)
  _Float16* hb1 = hb0 + 4 * TILE;             // h tile buffer 1      (4 KB)
  _Float16* xb0 = hb1 + 4 * TILE;             // x tile buffer 0      (2 KB)
  _Float16* xb1 = xb0 + 2 * TILE;             // x tile buffer 1      (2 KB)

  const int tid  = threadIdx.x;
  const int lane = tid & 31;
  const int wave = tid >> 5;
  const int grp  = blockIdx.x;
  const int row0 = grp * 16;       // batch rows of this WG
  const int ncol = wave * 16;      // hidden column base of this wave
  const int nlo  = lane & 15;

  // ---- Stage weights f32 -> f16, swizzled to B-fragment layout (once). ----
  for (int i = tid; i < F * G; i += 256) {
    const int k = i >> 9, col = i & (G - 1);           // W row-major [64][512]
    Wsw[((k >> 5) * 32 + (col >> 4)) * TILE + b_swz(k & 31, col & 15)] = (_Float16)W[i];
  }
  for (int i = tid; i < H * G; i += 256) {
    const int k = i >> 9, col = i & (G - 1);           // U row-major [128][512]
    Usw[((k >> 5) * 32 + (col >> 4)) * TILE + b_swz(k & 31, col & 15)] = (_Float16)U[i];
  }
  for (int i = tid; i < 4 * TILE; i += 256) hb0[i] = (_Float16)0.0f;

  // Stage x_0 into xb0.
  const int xm = tid >> 4, xq = (tid & 15) * 4;
  {
    const float4 xv = *(const float4*)&x[((size_t)(row0 + xm) * T) * F + xq];
    const float* xe = (const float*)&xv;
#pragma unroll
    for (int j = 0; j < 4; ++j) {
      const int f = xq + j;
      xb0[(f >> 5) * TILE + a_swz(xm, f & 31)] = (_Float16)xe[j];
    }
  }
  __syncthreads();

  // Hoisted, time-invariant per-lane constants.
  float bias[4];
#pragma unroll
  for (int g4 = 0; g4 < 4; ++g4) bias[g4] = b[g4 * H + ncol + nlo];

  const int hn    = ncol + nlo;                  // h write-back column
  const int h_kt  = hn >> 5;
  const int h_off = hn & 31;
  const int h_e   = (h_off & 7) + ((h_off & 16) ? 8 : 0);
  const int h_hlf = ((h_off >> 3) & 1) * 16;

  // Tile pointer for flat chain index i = kt*4 + g4 (i<8: x@W, else h@U).
  auto tp = [&](int i) -> const _Float16* {
    return (i < 8) ? (Wsw + ((i >> 2) * 32 + (i & 3) * 8 + wave) * TILE)
                   : (Usw + (((i - 8) >> 2) * 32 + (i & 3) * 8 + wave) * TILE);
  };

  v8f cfrag = {};  // cell-state tile [16x16] for this wave's hidden columns

  for (int t = 0; t < T; ++t) {
    _Float16* hcur  = (t & 1) ? hb1 : hb0;
    _Float16* hnext = (t & 1) ? hb0 : hb1;
    _Float16* xcur  = (t & 1) ? xb1 : xb0;
    _Float16* xnext = (t & 1) ? xb0 : xb1;

    // Issue next-step x load early so it hides under the WMMA chain.
    const int tn = (t + 1 < T) ? (t + 1) : t;
    const float4 xv = *(const float4*)&x[((size_t)(row0 + xm) * T + tn) * F + xq];

    if (t + 2 < T && tid < 32) {
      const int m = tid >> 1, c = (tid & 1) * 32;
      __builtin_prefetch(&x[((size_t)(row0 + m) * T + (t + 2)) * F + c], 0, 1);
    }

    // ---- A fragments (shared across this wave's 4 gate tiles). ----
    v16h xf[2], hf[4];
#pragma unroll
    for (int kt = 0; kt < 2; ++kt) xf[kt] = load_frag(xcur + kt * TILE, lane);
#pragma unroll
    for (int kt = 0; kt < 4; ++kt) hf[kt] = load_frag(hcur + kt * TILE, lane);

    // ---- 24-WMMA chain, B-fragments pipelined 2 deep. ----
    v8f zero = {};
    v8f acc[4];
    v16h bp0 = load_frag(tp(0), lane);
    v16h bp1 = load_frag(tp(1), lane);
#pragma unroll
    for (int i = 0; i < 24; ++i) {
      const v16h bu = bp0;
      bp0 = bp1;
      if (i + 2 < 24) bp1 = load_frag(tp(i + 2), lane);
      const v16h a = (i < 8) ? xf[i >> 2] : hf[(i - 8) >> 2];
      const int g4 = i & 3;
      acc[g4] = (i < 4) ? WMMA_F16(a, bu, zero) : WMMA_F16(a, bu, acc[g4]);
    }

    // ---- Stash x_{t+1} into the other x buffer. ----
    {
      const float* xe = (const float*)&xv;
#pragma unroll
      for (int j = 0; j < 4; ++j) {
        const int f = xq + j;
        xnext[(f >> 5) * TILE + a_swz(xm, f & 31)] = (_Float16)xe[j];
      }
    }

    // ---- Gate math in registers; write h tile to the other h buffer. ----
#pragma unroll
    for (int r = 0; r < 8; ++r) {
      const float iv = fast_sigmoid(acc[0][r] + bias[0]);
      const float fv = fast_sigmoid(acc[1][r] + bias[1]);
      const float gv = fast_tanh(acc[2][r] + bias[2]);
      const float ov = fast_sigmoid(acc[3][r] + bias[3]);
      const float cv = fv * cfrag[r] + iv * gv;
      cfrag[r] = cv;
      const float hv = ov * fast_tanh(cv);
      const int m = r + ((lane >> 4) << 3);
      hnext[h_kt * TILE + (m + h_hlf) * 16 + h_e] = (_Float16)hv;
    }
    __syncthreads();  // single barrier: h_t / x_{t+1} published to all waves

    // ---- Publish h_t: linear b128 copy LDS -> global tile block. ----
    {
      _Float16* dst = hs + ((size_t)grp * T + t) * (4 * TILE);
      *(v8h*)(dst + tid * 8) = *(const v8h*)(hnext + tid * 8);
    }
    // (hnext is next overwritten only after the barrier of step t+1)
  }
}

// ---------------------------------------------------------------------------
// Kernel 2: dense head  d1 = h@W1+b1 ; d2 = relu(d1@W2+b2) ;
//                       d3 = relu(d2@W3+b3) ; out = sigmoid(d3@W4+b4)
// grid = 1024, block = 256 (8 waves); wave handles one (group,t) tile of
// 16 batch rows. B-fragment loads pipelined 2 deep through each layer.
// ---------------------------------------------------------------------------
__global__ __launch_bounds__(256) void dense_kernel(
    const _Float16* __restrict__ hs,
    const float* __restrict__ W1, const float* __restrict__ b1,
    const float* __restrict__ W2, const float* __restrict__ b2,
    const float* __restrict__ W3, const float* __restrict__ b3,
    const float* __restrict__ W4, const float* __restrict__ b4,
    float* __restrict__ out) {
  extern __shared__ char smem[];
  _Float16* W1sw = (_Float16*)smem;           // 4 kt x 8 nt tiles (32 KB)
  _Float16* W2sw = W1sw + 4 * 8 * TILE;       // 4 kt x 8 nt tiles (32 KB)
  _Float16* W3sw = W2sw + 4 * 8 * TILE;       // 4 kt x 2 nt tiles (8 KB)
  _Float16* stage = W3sw + 4 * 2 * TILE;      // 8 waves x 4 kt tiles (32 KB)
  float* b1s = (float*)(stage + 8 * 4 * TILE);
  float* b2s = b1s + 128;
  float* b3s = b2s + 128;
  float* w4s = b3s + 32;
  float* d3s = w4s + 32;                      // 8 waves x [16][32] f32

  const int tid  = threadIdx.x;
  const int lane = tid & 31;
  const int wave = tid >> 5;
  const int nlo  = lane & 15;

  // ---- Stage weights, swizzled to B-fragment layout. ----
  for (int i = tid; i < H * 128; i += 256) {
    const int k = i >> 7, col = i & 127;
    const int p = ((k >> 5) * 8 + (col >> 4)) * TILE + b_swz(k & 31, col & 15);
    W1sw[p] = (_Float16)W1[i];
    W2sw[p] = (_Float16)W2[i];
  }
  for (int i = tid; i < H * 32; i += 256) {
    const int k = i >> 5, col = i & 31;
    W3sw[((k >> 5) * 2 + (col >> 4)) * TILE + b_swz(k & 31, col & 15)] = (_Float16)W3[i];
  }
  for (int i = tid; i < 128; i += 256) { b1s[i] = b1[i]; b2s[i] = b2[i]; }
  if (tid < 32) { b3s[tid] = b3[tid]; w4s[tid] = W4[tid]; }
  __syncthreads();

  const int tileIdx = blockIdx.x * 8 + wave;  // 0 .. 4*T-1
  const int grp = tileIdx >> 11;              // /T
  const int t   = tileIdx & (T - 1);
  _Float16* st = stage + wave * 4 * TILE;
  float*    d3 = d3s + wave * 16 * 32;

  v8f zero = {};

  // ---- h A-fragments: contiguous b128 global loads from tile blocks. ----
  const _Float16* hblk = hs + (size_t)tileIdx * (4 * TILE);
  v16h hfrag[4];
#pragma unroll
  for (int kt = 0; kt < 4; ++kt) hfrag[kt] = load_frag(hblk + kt * TILE, lane);

  // ---- d1 = h@W1 + b1 (linear) -> stage (A-swizzled f16). ----
  {
    auto tp = [&](int i) { return W1sw + ((i & 3) * 8 + (i >> 2)) * TILE; };
    v16h bp0 = load_frag(tp(0), lane);
    v16h bp1 = load_frag(tp(1), lane);
#pragma unroll
    for (int nt = 0; nt < 8; ++nt) {
      v8f acc;
#pragma unroll
      for (int kt = 0; kt < 4; ++kt) {
        const int i = nt * 4 + kt;
        const v16h bu = bp0;
        bp0 = bp1;
        if (i + 2 < 32) bp1 = load_frag(tp(i + 2), lane);
        acc = (kt == 0) ? WMMA_F16(hfrag[0], bu, zero)
                        : WMMA_F16(hfrag[kt], bu, acc);
      }
      const float bv = b1s[nt * 16 + nlo];
      const int col = nt * 16 + nlo, kt2 = col >> 5, koff = col & 31;
#pragma unroll
      for (int r = 0; r < 8; ++r) {
        const int m = r + ((lane >> 4) << 3);
        st[kt2 * TILE + a_swz(m, koff)] = (_Float16)(acc[r] + bv);
      }
    }
  }

  // ---- d2 = relu(d1@W2 + b2) -> stage (A-swizzled f16). ----
  v16h dfrag[4];
#pragma unroll
  for (int kt = 0; kt < 4; ++kt) dfrag[kt] = load_frag(st + kt * TILE, lane);
  {
    auto tp = [&](int i) { return W2sw + ((i & 3) * 8 + (i >> 2)) * TILE; };
    v16h bp0 = load_frag(tp(0), lane);
    v16h bp1 = load_frag(tp(1), lane);
#pragma unroll
    for (int nt = 0; nt < 8; ++nt) {
      v8f acc;
#pragma unroll
      for (int kt = 0; kt < 4; ++kt) {
        const int i = nt * 4 + kt;
        const v16h bu = bp0;
        bp0 = bp1;
        if (i + 2 < 32) bp1 = load_frag(tp(i + 2), lane);
        acc = (kt == 0) ? WMMA_F16(dfrag[0], bu, zero)
                        : WMMA_F16(dfrag[kt], bu, acc);
      }
      const float bv = b2s[nt * 16 + nlo];
      const int col = nt * 16 + nlo, kt2 = col >> 5, koff = col & 31;
#pragma unroll
      for (int r = 0; r < 8; ++r) {
        const int m = r + ((lane >> 4) << 3);
        const float v = acc[r] + bv;
        st[kt2 * TILE + a_swz(m, koff)] = (_Float16)(v > 0.0f ? v : 0.0f);
      }
    }
  }

  // ---- d3 = relu(d2@W3 + b3) -> d3s (f32 [16][32]). ----
#pragma unroll
  for (int kt = 0; kt < 4; ++kt) dfrag[kt] = load_frag(st + kt * TILE, lane);
  {
    auto tp = [&](int i) { return W3sw + ((i & 3) * 2 + (i >> 2)) * TILE; };
    v16h bp0 = load_frag(tp(0), lane);
    v16h bp1 = load_frag(tp(1), lane);
#pragma unroll
    for (int nt = 0; nt < 2; ++nt) {
      v8f acc;
#pragma unroll
      for (int kt = 0; kt < 4; ++kt) {
        const int i = nt * 4 + kt;
        const v16h bu = bp0;
        bp0 = bp1;
        if (i + 2 < 8) bp1 = load_frag(tp(i + 2), lane);
        acc = (kt == 0) ? WMMA_F16(dfrag[0], bu, zero)
                        : WMMA_F16(dfrag[kt], bu, acc);
      }
      const float bv = b3s[nt * 16 + nlo];
#pragma unroll
      for (int r = 0; r < 8; ++r) {
        const int m = r + ((lane >> 4) << 3);
        const float v = acc[r] + bv;
        d3[m * 32 + nt * 16 + nlo] = (v > 0.0f ? v : 0.0f);
      }
    }
  }

  // ---- out = sigmoid(d3 @ W4 + b4): one lane per batch row. ----
  const float b4v = b4[0];
  if (lane < 16) {
    float s = 0.0f;
#pragma unroll
    for (int j = 0; j < 32; ++j) s += d3[lane * 32 + j] * w4s[j];
    out[((size_t)(grp * 16 + lane)) * T + t] = fast_sigmoid(s + b4v);
  }
}

extern "C" void kernel_launch(void* const* d_in, const int* in_sizes, int n_in,
                              void* d_out, int out_size, void* d_ws, size_t ws_size,
                              hipStream_t stream) {
  const float* x  = (const float*)d_in[0];
  const float* W  = (const float*)d_in[1];
  const float* U  = (const float*)d_in[2];
  const float* b  = (const float*)d_in[3];
  const float* W1 = (const float*)d_in[4];
  const float* b1 = (const float*)d_in[5];
  const float* W2 = (const float*)d_in[6];
  const float* b2 = (const float*)d_in[7];
  const float* W3 = (const float*)d_in[8];
  const float* b3 = (const float*)d_in[9];
  const float* W4 = (const float*)d_in[10];
  const float* b4 = (const float*)d_in[11];
  float* out = (float*)d_out;

  _Float16* hs = (_Float16*)d_ws;  // [4][T][4*TILE] f16 tile blocks, 33.5 MB

  constexpr size_t lds1 =
      (size_t)(2 * 32 * TILE + 4 * 32 * TILE + 2 * 4 * TILE + 2 * 2 * TILE) * sizeof(_Float16);
  constexpr size_t lds2 =
      (size_t)(4 * 8 * TILE * 2 + 4 * 2 * TILE + 8 * 4 * TILE) * sizeof(_Float16) +
      (size_t)(128 + 128 + 32 + 32 + 8 * 16 * 32) * sizeof(float);

  lstm_kernel<<<Bsz / 16, 256, lds1, stream>>>(x, W, U, b, hs);
  dense_kernel<<<(Bsz * T) / (16 * 8), 256, lds2, stream>>>(
      hs, W1, b1, W2, b2, W3, b3, W4, b4, out);
}
